// F_func_hz_RNN_54709293416517
// MI455X (gfx1250) — matrix operand order repaired and checked by reading, not compiled, not run on hardware
//
#include <hip/hip_runtime.h>
#include <cstdint>
#include <cstddef>

#define NODE   207
#define HID    16
#define BATCH  64
#define TSTEPS 12
#define HD     6624          // HIDDEN_DIM = NODE*HID*2
#define INPD   16560         // NODE*HID*5
#define KPAD   16576         // pad to multiple of 32 (518*32)
#define KSTEPS 518
#define NTILES 414           // HD/16
#define SPLITK 7             // 518 = 7 * 74
#define KSPW   74            // K-steps per wave
#define PFDIST (16 * 32)     // prefetch 16 K-steps (1KB) ahead in the B stream

typedef unsigned short u16;
typedef __attribute__((ext_vector_type(16))) __bf16 v16bf;
typedef __attribute__((ext_vector_type(8)))  __bf16 v8bf;
typedef __attribute__((ext_vector_type(8)))  float  v8f;

__device__ __forceinline__ u16 f2bf(float x) {
  unsigned u = __float_as_uint(x);
  u += 0x7FFFu + ((u >> 16) & 1u);          // round-to-nearest-even
  return (u16)(u >> 16);
}
__device__ __forceinline__ float bf2f(u16 x) {
  return __uint_as_float(((unsigned)x) << 16);
}

// 32 contiguous bytes -> v16bf (two b128 loads)
__device__ __forceinline__ v16bf ld_contig32(const u16* p) {
  v8bf lo = *(const v8bf*)p;
  v8bf hi = *(const v8bf*)(p + 8);
  return __builtin_shufflevector(lo, hi, 0,1,2,3,4,5,6,7,8,9,10,11,12,13,14,15);
}
// A 16x32 fragment: per-lane K groups {kb+hi*8 .. +7} and {kb+16+hi*8 .. +7}
__device__ __forceinline__ v16bf ld_fragA(const u16* row, int kb, int hi) {
  v8bf lo = *(const v8bf*)(row + kb + hi * 8);
  v8bf hg = *(const v8bf*)(row + kb + 16 + hi * 8);
  return __builtin_shufflevector(lo, hg, 0,1,2,3,4,5,6,7,8,9,10,11,12,13,14,15);
}

// ---------------- weight conversion (fp32 -> bf16, once per launch) --------
__global__ void cvt_wih(const float* __restrict__ wih, u16* __restrict__ out) {
  size_t tid = (size_t)blockIdx.x * blockDim.x + threadIdx.x;
  size_t total = (size_t)HD * KPAD;
  if (tid >= total) return;
  int k = (int)(tid % KPAD);
  size_t n = tid / KPAD;
  float v = (k < INPD) ? wih[n * (size_t)INPD + k] : 0.0f;
  out[tid] = f2bf(v);
}

__global__ void cvt_whh(const float* __restrict__ whh, u16* __restrict__ out) {
  size_t tid = (size_t)blockIdx.x * blockDim.x + threadIdx.x;
  size_t total = (size_t)HD * HD;
  if (tid >= total) return;
  out[tid] = f2bf(whh[tid]);
}

__global__ void zero_f32(float* __restrict__ p, int n) {
  int i = blockIdx.x * blockDim.x + threadIdx.x;
  if (i < n) p[i] = 0.0f;
}

// ---------------- per-timestep prep: f / g / fg / 1x1-conv -> bf16 inp -----
// one wave per (b,n); 8 waves / block; 1656 blocks.
__global__ void __launch_bounds__(256)
prep_kernel(const float* __restrict__ hz_in, const float* __restrict__ spline,
            int t,
            const float* __restrict__ Wf, const float* __restrict__ bfv,
            const float* __restrict__ Wg, const float* __restrict__ bgv,
            const float* __restrict__ convw,
            u16* __restrict__ inp) {
  __shared__ float s_hz[8][32];
  __shared__ float s_f [8][32];
  __shared__ float s_g [8][256];
  int wave = threadIdx.x >> 5;
  int lane = threadIdx.x & 31;
  int p = blockIdx.x * 8 + wave;
  int b = p / NODE;
  int n = p % NODE;

  s_hz[wave][lane] = hz_in[(size_t)b * HD + n * 32 + lane];
  __syncthreads();

  // f_vec[lane] = tanh(bf[lane] + sum_i h_i * Wf[i][lane])   (h_i = hz[2i])
  float a = bfv[lane];
  #pragma unroll
  for (int i = 0; i < 16; ++i) a += s_hz[wave][2 * i] * Wf[i * 32 + lane];
  s_f[wave][lane] = tanhf(a);

  // g: 8 outputs per lane, o = q*32+lane; z_k = hz[2k+1]
  #pragma unroll
  for (int q = 0; q < 8; ++q) {
    int o = q * 32 + lane;
    float g = bgv[o];
    #pragma unroll
    for (int k = 0; k < 16; ++k) g += s_hz[wave][2 * k + 1] * Wg[k * 256 + o];
    s_g[wave][o] = tanhf(g);
  }
  __syncthreads();

  // fg[i][c] = sum_j g[i][j] * f_vec[2j+c], lane = i*2+c
  int i = lane >> 1, c = lane & 1;
  float fg = 0.0f;
  #pragma unroll
  for (int j = 0; j < 16; ++j) fg += s_g[wave][i * 16 + j] * s_f[wave][2 * j + c];

  u16* row = inp + (size_t)b * KPAD;
  int base = 3312 + n * 64 + i * 4 + c * 2;   // hz4 index ((n*16+i)*2+c)*2+s
  row[base]     = f2bf(s_f[wave][lane]);      // s=0 -> f
  row[base + 1] = f2bf(fg);                   // s=1 -> fg
  if (lane < 16) {                            // x part: conv_w[h]*v at h*207+n
    float v = spline[(((size_t)t * BATCH + b) * NODE + n) * 2 + 1];
    row[lane * NODE + n] = f2bf(convw[lane] * v);
  }
  if (n == 0 && lane < 16) row[INPD + lane] = (u16)0;  // K padding
}

// ---------------- big GEMM: split-K bf16 WMMA -------------------------------
// prepart[split] (64x6624) partial of inp(64xKPAD) * Wih^T over K-range.
// one wave per (ntile, split); 4 M-tiles per wave; 4 waves/block.
// 414*7 = 2898 waves -> 725 blocks: enough MLP to saturate 23.3 TB/s HBM.
__global__ void __launch_bounds__(128)
gemm_wmma(const u16* __restrict__ A, const u16* __restrict__ Bw,
          float* __restrict__ prepart) {
  int w = blockIdx.x * 4 + (threadIdx.x >> 5);
  if (w >= NTILES * SPLITK) return;
  int lane  = threadIdx.x & 31;
  int split = w % SPLITK;
  int ntile = w / SPLITK;
  int l15 = lane & 15;
  int hi  = lane >> 4;

  const u16* brow = Bw + (size_t)(ntile * 16 + l15) * KPAD;  // B[k][n]=Wih[n][k]
  const u16* ar0  = A + (size_t)l15 * KPAD;                  // m-tile 0 row
  const u16* ar1  = ar0 + (size_t)16 * KPAD;
  const u16* ar2  = ar0 + (size_t)32 * KPAD;
  const u16* ar3  = ar0 + (size_t)48 * KPAD;

  v8f acc0 = {}, acc1 = {}, acc2 = {}, acc3 = {};

  int ks0 = split * KSPW;
  for (int ks = ks0; ks < ks0 + KSPW; ++ks) {
    int kb = ks * 32;
    v16bf bv = ld_fragA(brow, kb, hi);  // B 32x16 frag
    v16bf a0 = ld_fragA(ar0, kb, hi);
    v16bf a1 = ld_fragA(ar1, kb, hi);
    v16bf a2 = ld_fragA(ar2, kb, hi);
    v16bf a3 = ld_fragA(ar3, kb, hi);
    __builtin_prefetch((const void*)(brow + kb + PFDIST + hi * 16), 0, 3);

    acc0 = __builtin_amdgcn_wmma_f32_16x16x32_bf16(false, a0, false, bv,
                                                   (short)0, acc0, false, false);
    acc1 = __builtin_amdgcn_wmma_f32_16x16x32_bf16(false, a1, false, bv,
                                                   (short)0, acc1, false, false);
    acc2 = __builtin_amdgcn_wmma_f32_16x16x32_bf16(false, a2, false, bv,
                                                   (short)0, acc2, false, false);
    acc3 = __builtin_amdgcn_wmma_f32_16x16x32_bf16(false, a3, false, bv,
                                                   (short)0, acc3, false, false);
  }

  int ncol = ntile * 16 + l15;
  float* part = prepart + (size_t)split * BATCH * HD;
  #pragma unroll
  for (int mt = 0; mt < 4; ++mt) {
    const v8f& acc = (mt == 0 ? acc0 : mt == 1 ? acc1 : mt == 2 ? acc2 : acc3);
    #pragma unroll
    for (int r = 0; r < 8; ++r) {                  // C/D layout: m = r + hi*8
      int m = mt * 16 + hi * 8 + r;
      part[(size_t)m * HD + ncol] = acc[r];
    }
  }
}

// fold split-K partials + b_ih -> pre
__global__ void __launch_bounds__(256)
reduce_pre(const float* __restrict__ part, const float* __restrict__ bih,
           float* __restrict__ pre) {
  int i = blockIdx.x * blockDim.x + threadIdx.x;   // over BATCH*HD
  if (i >= BATCH * HD) return;
  float s = bih[i % HD];
  #pragma unroll
  for (int sp = 0; sp < SPLITK; ++sp) s += part[(size_t)sp * BATCH * HD + i];
  pre[i] = s;
}

// ---------------- sequential RNN step: out = tanh(pre + Whh*h + bhh) --------
// one wave per output row; Whh bf16 stays L2-resident (87.8 MB < 192 MB).
__global__ void __launch_bounds__(256)
matvec_tanh(const u16* __restrict__ Whh, const float* __restrict__ hprev,
            const float* __restrict__ prer, const float* __restrict__ bhh,
            float* __restrict__ out) {
  int wave = threadIdx.x >> 5;
  int lane = threadIdx.x & 31;
  int row  = blockIdx.x * 8 + wave;
  const u16* wrow = Whh + (size_t)row * HD;

  float sum = 0.0f;
  for (int c = lane; c < HD / 4; c += 32) {
    int k = c * 4;
    uint2  w4 = *(const uint2*)(wrow + k);
    float4 h4 = *(const float4*)(hprev + k);
    sum += bf2f((u16)(w4.x & 0xFFFFu)) * h4.x;
    sum += bf2f((u16)(w4.x >> 16))     * h4.y;
    sum += bf2f((u16)(w4.y & 0xFFFFu)) * h4.z;
    sum += bf2f((u16)(w4.y >> 16))     * h4.w;
  }
  #pragma unroll
  for (int off = 16; off >= 1; off >>= 1) sum += __shfl_xor(sum, off, 32);
  if (lane == 0) out[row] = tanhf(sum + prer[row] + bhh[row]);
}

// ---------------------------------------------------------------------------
extern "C" void kernel_launch(void* const* d_in, const int* in_sizes, int n_in,
                              void* d_out, int out_size, void* d_ws, size_t ws_size,
                              hipStream_t stream) {
  const float* spline = (const float*)d_in[0];
  const float* init0  = (const float*)d_in[1];
  const float* Wf     = (const float*)d_in[2];
  const float* bfv    = (const float*)d_in[3];
  const float* Wg     = (const float*)d_in[4];
  const float* bgv    = (const float*)d_in[5];
  const float* convw  = (const float*)d_in[6];
  const float* Wih    = (const float*)d_in[7];
  const float* bih    = (const float*)d_in[8];
  const float* Whh    = (const float*)d_in[9];
  const float* bhh    = (const float*)d_in[10];

  char* ws = (char*)d_ws;
  size_t off = 0;
  auto take = [&](size_t bytes) {
    char* p = ws + off;
    off = (off + bytes + 255) & ~(size_t)255;
    return p;
  };
  u16*   wih_bf  = (u16*)  take((size_t)HD * KPAD * 2);            // 219.6 MB
  u16*   whh_bf  = (u16*)  take((size_t)HD * HD * 2);              //  87.8 MB
  u16*   inp_bf  = (u16*)  take((size_t)BATCH * KPAD * 2);         //   2.1 MB
  float* prepart = (float*)take((size_t)SPLITK * BATCH * HD * 4);  //  11.9 MB
  float* pre     = (float*)take((size_t)BATCH * HD * 4);
  float* hz0     = (float*)take((size_t)BATCH * HD * 4);
  float* hz1     = (float*)take((size_t)BATCH * HD * 4);
  float* zrow    = (float*)take((size_t)HD * 4);

  {
    size_t tot = (size_t)HD * KPAD;
    cvt_wih<<<(unsigned)((tot + 255) / 256), 256, 0, stream>>>(Wih, wih_bf);
  }
  {
    size_t tot = (size_t)HD * HD;
    cvt_whh<<<(unsigned)((tot + 255) / 256), 256, 0, stream>>>(Whh, whh_bf);
  }
  zero_f32<<<(HD + 255) / 256, 256, 0, stream>>>(zrow, HD);

  float* hzbuf[2] = { hz0, hz1 };
  for (int t = 0; t < TSTEPS; ++t) {
    const float* hz_in = (t == 0) ? init0 : hzbuf[(t + 1) & 1];
    float* hz_outb = (t == TSTEPS - 1) ? (float*)d_out : hzbuf[t & 1];

    prep_kernel<<<(BATCH * NODE) / 8, 256, 0, stream>>>(
        hz_in, spline, t, Wf, bfv, Wg, bgv, convw, inp_bf);

    gemm_wmma<<<(NTILES * SPLITK + 3) / 4, 128, 0, stream>>>(
        inp_bf, wih_bf, prepart);
    reduce_pre<<<(BATCH * HD + 255) / 256, 256, 0, stream>>>(prepart, bih, pre);

    for (int b = 0; b < BATCH; ++b) {
      const float* hp = (b == 0) ? zrow : (hz_outb + (size_t)(b - 1) * HD);
      matvec_tanh<<<HD / 8, 256, 0, stream>>>(
          whh_bf, hp, pre + (size_t)b * HD, bhh, hz_outb + (size_t)b * HD);
    }
  }
  (void)in_sizes; (void)n_in; (void)out_size; (void)ws_size;
}
